// LinearSDE_76501957476837
// MI455X (gfx1250) — matrix operand order repaired and checked by reading, not compiled, not run on hardware
//
#include <hip/hip_runtime.h>

typedef __attribute__((ext_vector_type(16))) __bf16 v16bf;
typedef __attribute__((ext_vector_type(8)))  __bf16 bf16x8;
typedef __attribute__((ext_vector_type(8)))  float  v8f;

#define BPATHS 2048
#define DDIM   64
#define NSTEPS 500
#define DT     0.002f
#define SQRTDT 0.04472135954999579f

#define XP 72   // bf16 row stride in LDS: 144B rows -> 16B aligned, bank-spread (gcd(36,64)=4)

__device__ __forceinline__ __bf16 f2bf(float f) { return (__bf16)f; }

__global__ __launch_bounds__(128)
void sde_em_kernel(const float* __restrict__ drift,
                   const float* __restrict__ drift_bias,
                   const float* __restrict__ diffusion,
                   const float* __restrict__ diffusion_bias,
                   const float* __restrict__ x0,
                   const float* __restrict__ noise,
                   float* __restrict__ out)
{
    // double-buffered bf16 hi/lo mirror of the 16x64 state tile (fp32 master in regs)
    __shared__ __align__(16) __bf16 xhi[2][16][XP];
    __shared__ __align__(16) __bf16 xlo[2][16][XP];

    const int tid  = threadIdx.x;
    const int lane = tid & 31;
    const int jt   = tid >> 5;      // wave id == column tile (wave owns 16 output dims)
    const int h    = lane >> 4;     // lane half
    const int col  = lane & 15;     // B-layout column / A-layout row / C-layout column
    const int n    = jt * 16 + col; // this lane's output dim
    const int b0   = blockIdx.x * 16;

    // ---- one-time: B fragments in registers, pre-scaled by dt / sqrt(dt) ----
    // B layout used here (32x16 bf16): lane col N = lane%16, K = 16h + e contiguous (same as A).
    // bf16x3 split of the *scaled* fp32 weight keeps ~17 mantissa bits.
    v16bf wFh[2], wFl[2], wGh[2], wGl[2];
    #pragma unroll
    for (int kt = 0; kt < 2; ++kt) {
        int kb = 16 * h + 32 * kt;
        #pragma unroll
        for (int e = 0; e < 16; ++e) {
            float a = drift[n * DDIM + kb + e] * DT;
            __bf16 ah = f2bf(a);
            wFh[kt][e] = ah;
            wFl[kt][e] = f2bf(a - (float)ah);
            float c = diffusion[n * DDIM + kb + e] * SQRTDT;
            __bf16 ch = f2bf(c);
            wGh[kt][e] = ch;
            wGl[kt][e] = f2bf(c - (float)ch);
        }
    }
    const float fbv = drift_bias[n] * DT;        // folded into accumulator init
    const float gbv = diffusion_bias[n] * SQRTDT;

    // ---- init: fp32 state in C-layout registers + bf16 hi/lo LDS mirror + out[0] ----
    v8f xreg;
    #pragma unroll
    for (int r = 0; r < 8; ++r) {
        int M = r + 8 * h;
        float v = x0[(size_t)(b0 + M) * DDIM + n];
        xreg[r] = v;
        out[(size_t)(b0 + M) * DDIM + n] = v;
        __bf16 hi = f2bf(v);
        xhi[0][M][n] = hi;
        xlo[0][M][n] = f2bf(v - (float)hi);
    }
    __syncthreads();

    // ---- running per-lane pointers; per-row deltas become load/store immediates ----
    // noise[(b0+8h+r)*NSTEPS*DDIM + s*DDIM + n]: row delta r*NSTEPS*DDIM*4 = r*128000 B (imm),
    // step delta DDIM floats. out[(s+1)*B*D + (b0+8h+r)*D + n]: row delta r*256 B (imm),
    // step delta B*D floats.
    const float* np = noise + (size_t)(b0 + 8 * h) * NSTEPS * DDIM + n;
    float*       op = out + (size_t)BPATHS * DDIM + (size_t)(b0 + 8 * h) * DDIM + n;

    for (int s = 0; s < NSTEPS; ++s) {
        const int cur = s & 1, nxt = cur ^ 1;

        // prefetch next step's noise rows (global_prefetch_b8); offsets are immediates
        if (s + 1 < NSTEPS) {
            #pragma unroll
            for (int r = 0; r < 8; ++r) {
                __builtin_prefetch(np + (size_t)r * NSTEPS * DDIM + DDIM, 0, 1);
            }
        }

        // ---- A fragments: pure contiguous 16B LDS loads, zero conversion VALU ----
        // A layout (16x32 bf16): lane row M = lane%16; element e -> K=16h+e,
        // matching B; loaded as two contiguous 8x bf16 runs.
        v16bf ahi[2], alo[2];
        #pragma unroll
        for (int kt = 0; kt < 2; ++kt) {
            bf16x8 h0 = *(const bf16x8*)&xhi[cur][col][16 * h + 32 * kt];
            bf16x8 h1 = *(const bf16x8*)&xhi[cur][col][8 + 16 * h + 32 * kt];
            bf16x8 l0 = *(const bf16x8*)&xlo[cur][col][16 * h + 32 * kt];
            bf16x8 l1 = *(const bf16x8*)&xlo[cur][col][8 + 16 * h + 32 * kt];
            ahi[kt] = __builtin_shufflevector(h0, h1, 0,1,2,3,4,5,6,7,8,9,10,11,12,13,14,15);
            alo[kt] = __builtin_shufflevector(l0, l1, 0,1,2,3,4,5,6,7,8,9,10,11,12,13,14,15);
        }

        // ---- this step's noise: one base register, 8 immediate-offset loads ----
        float nz[8];
        #pragma unroll
        for (int r = 0; r < 8; ++r) {
            nz[r] = np[(size_t)r * NSTEPS * DDIM];
        }

        // ---- four independent 3-deep WMMA chains (per k-tile, per output) ----
        v8f fa0, ga0;
        #pragma unroll
        for (int i = 0; i < 8; ++i) { fa0[i] = fbv; ga0[i] = gbv; }
        v8f z = {0.f, 0.f, 0.f, 0.f, 0.f, 0.f, 0.f, 0.f};

        v8f fa1, ga1;
        fa0 = __builtin_amdgcn_wmma_f32_16x16x32_bf16(false, ahi[0], false, wFh[0], (short)0, fa0, false, false);
        ga0 = __builtin_amdgcn_wmma_f32_16x16x32_bf16(false, ahi[0], false, wGh[0], (short)0, ga0, false, false);
        fa1 = __builtin_amdgcn_wmma_f32_16x16x32_bf16(false, ahi[1], false, wFh[1], (short)0, z,   false, false);
        ga1 = __builtin_amdgcn_wmma_f32_16x16x32_bf16(false, ahi[1], false, wGh[1], (short)0, z,   false, false);
        fa0 = __builtin_amdgcn_wmma_f32_16x16x32_bf16(false, alo[0], false, wFh[0], (short)0, fa0, false, false);
        ga0 = __builtin_amdgcn_wmma_f32_16x16x32_bf16(false, alo[0], false, wGh[0], (short)0, ga0, false, false);
        fa1 = __builtin_amdgcn_wmma_f32_16x16x32_bf16(false, alo[1], false, wFh[1], (short)0, fa1, false, false);
        ga1 = __builtin_amdgcn_wmma_f32_16x16x32_bf16(false, alo[1], false, wGh[1], (short)0, ga1, false, false);
        fa0 = __builtin_amdgcn_wmma_f32_16x16x32_bf16(false, ahi[0], false, wFl[0], (short)0, fa0, false, false);
        ga0 = __builtin_amdgcn_wmma_f32_16x16x32_bf16(false, ahi[0], false, wGl[0], (short)0, ga0, false, false);
        fa1 = __builtin_amdgcn_wmma_f32_16x16x32_bf16(false, ahi[1], false, wFl[1], (short)0, fa1, false, false);
        ga1 = __builtin_amdgcn_wmma_f32_16x16x32_bf16(false, ahi[1], false, wGl[1], (short)0, ga1, false, false);

        // ---- update: xn = x + f*dt + g*dW with all scalars pre-folded ----
        #pragma unroll
        for (int r = 0; r < 8; ++r) {
            int M = r + 8 * h;
            float xn = xreg[r] + (fa0[r] + fa1[r]) + (ga0[r] + ga1[r]) * nz[r];
            xreg[r] = xn;
            op[(size_t)r * DDIM] = xn;           // immediate offset r*256B
            __bf16 hi = f2bf(xn);
            xhi[nxt][M][n] = hi;
            xlo[nxt][M][n] = f2bf(xn - (float)hi);
        }

        np += DDIM;                  // one u64 add per step
        op += (size_t)BPATHS * DDIM; // one u64 add per step
        __syncthreads();             // writes to buf nxt visible before next step's A build
    }
}

extern "C" void kernel_launch(void* const* d_in, const int* in_sizes, int n_in,
                              void* d_out, int out_size, void* d_ws, size_t ws_size,
                              hipStream_t stream) {
    const float* drift  = (const float*)d_in[0];
    const float* dbias  = (const float*)d_in[1];
    const float* diffu  = (const float*)d_in[2];
    const float* gbias  = (const float*)d_in[3];
    const float* x0     = (const float*)d_in[4];
    const float* noise  = (const float*)d_in[5];
    float* out = (float*)d_out;

    dim3 grid(BPATHS / 16);   // 128 workgroups: one 16-path tile each
    dim3 block(128);          // 4 waves: one column tile per wave
    hipLaunchKernelGGL(sde_em_kernel, grid, block, 0, stream,
                       drift, dbias, diffu, gbias, x0, noise, out);
}
